// QuantizedCyclicalAttention_68032281969134
// MI455X (gfx1250) — matrix operand order repaired and checked by reading, not compiled, not run on hardware
//
#include <hip/hip_runtime.h>
#include <hip/hip_bf16.h>
#include <math.h>

// ---------------------------------------------------------------------------
// QuantizedCyclicalAttention for MI455X (gfx1250), wave32 + WMMA f16 + TDM.
// B=1, S=2048, D=1024, H=16, HD=64, TF=32, KEY_BITS=4, VALUE_BITS=8
// ---------------------------------------------------------------------------

#define S_LEN 2048
#define D_DIM 1024
#define NHEAD 16
#define HDIM  64
#define TFDIM 32

typedef __attribute__((ext_vector_type(16))) _Float16 v16h;
typedef __attribute__((ext_vector_type(8)))  _Float16 v8h;
typedef __attribute__((ext_vector_type(8)))  float    v8f;

union V16HU { v16h v; v8h h[2]; };

// ---------------- TDM availability probe -----------------------------------
#if defined(__has_builtin)
#if __has_builtin(__builtin_amdgcn_tensor_load_to_lds) && \
    __has_builtin(__builtin_amdgcn_s_wait_tensorcnt)
#define HAVE_TDM 1
#endif
#endif
#ifndef HAVE_TDM
#define HAVE_TDM 0
#endif

#if HAVE_TDM
typedef unsigned int u32x4 __attribute__((ext_vector_type(4)));
typedef int          i32x4 __attribute__((ext_vector_type(4)));
typedef int          i32x8 __attribute__((ext_vector_type(8)));

// Issue a 2-D TENSOR_LOAD_TO_LDS: tile (tile_w elems x tile_h rows) of 2-byte
// elements starting at gaddr, row stride = row_stride elems, into LDS offset
// lds_off. Descriptor per CDNA5 ISA 8.3/8.4 (count=1, type=2, data_size=2B).
__device__ __forceinline__ void tdm_load_2d_f16(const void* gaddr,
                                                unsigned lds_off,
                                                unsigned tile_w, unsigned tile_h,
                                                unsigned tensor_w,
                                                unsigned tensor_h,
                                                unsigned row_stride) {
  unsigned long long ga = (unsigned long long)(uintptr_t)gaddr;
  u32x4 g0;
  g0[0] = 1u;                                      // count=1 (valid user D#)
  g0[1] = lds_off;                                 // lds_addr (bytes)
  g0[2] = (unsigned)(ga & 0xffffffffull);          // global_addr[31:0]
  g0[3] = (unsigned)((ga >> 32) & 0x01ffffffull)   // global_addr[56:32]
          | (2u << 30);                            // type=2 ("image")
  i32x8 g1;
  g1[0] = (int)(1u << 16);                         // wg_mask=0, data_size=1(2B)
  g1[1] = (int)((tensor_w & 0xffffu) << 16);       // tensor_dim0[15:0]
  g1[2] = (int)(((tensor_w >> 16) & 0xffffu) |     // tensor_dim0[31:16]
                ((tensor_h & 0xffffu) << 16));     // tensor_dim1[15:0]
  g1[3] = (int)(((tensor_h >> 16) & 0xffffu) |     // tensor_dim1[31:16]
                ((tile_w & 0xffffu) << 16));       // tile_dim0
  g1[4] = (int)(tile_h & 0xffffu);                 // tile_dim1 (tile_dim2=0)
  g1[5] = (int)row_stride;                         // tensor_dim0_stride[31:0]
  g1[6] = 0;                                       // stride hi / dim1_stride
  g1[7] = 0;
  i32x4 z4 = {0, 0, 0, 0};
#if __clang_major__ >= 23
  i32x8 z8 = {0, 0, 0, 0, 0, 0, 0, 0};
  __builtin_amdgcn_tensor_load_to_lds(g0, g1, z4, z4, z8, 0);
#else
  __builtin_amdgcn_tensor_load_to_lds(g0, g1, z4, z4, 0);
#endif
}
#endif  // HAVE_TDM

// ---- WMMA fragment loaders (per CDNA5 ISA 7.12.2 layouts) -----------------
// A (16x32 f16, MxK): lane L holds row M=L&15.
//   lanes 0-15 : halves 0-7 -> K = k0+0..7,  halves 8-15 -> K = k0+16..23
//   lanes 16-31: halves 0-7 -> K = k0+8..15, halves 8-15 -> K = k0+24..31
__device__ __forceinline__ v16h load_frag_A(const _Float16* base, int ldh,
                                            int row_base, int k0) {
  int lane = threadIdx.x & 31;
  int half = lane >> 4;
  int row  = row_base + (lane & 15);
  const _Float16* p = base + (size_t)row * ldh + k0 + half * 8;
  V16HU r;
  r.h[0] = *(const v8h*)(p);
  r.h[1] = *(const v8h*)(p + 16);
  return r.v;
}

// B (32x16 f16, KxN) read from Bt[N][K] storage: lane L holds column N=L&15.
//   lanes 0-15 : K = k0+0..7, k0+16..23; lanes 16-31: K = k0+8..15, k0+24..31
__device__ __forceinline__ v16h load_frag_B(const _Float16* base, int ldh,
                                            int n_base, int k0) {
  int lane = threadIdx.x & 31;
  int half = lane >> 4;
  int n    = n_base + (lane & 15);
  const _Float16* p = base + (size_t)n * ldh + k0 + half * 8;
  V16HU r;
  r.h[0] = *(const v8h*)(p);
  r.h[1] = *(const v8h*)(p + 16);
  return r.v;
}

// ---------------------------------------------------------------------------
// f32 -> f16 elementwise convert
// ---------------------------------------------------------------------------
__global__ void k_f32_to_f16(const float* __restrict__ in,
                             _Float16* __restrict__ out, int n) {
  for (int i = blockIdx.x * blockDim.x + threadIdx.x; i < n;
       i += gridDim.x * blockDim.x)
    out[i] = (_Float16)in[i];
}

// ---------------------------------------------------------------------------
// W[K][N] f32 -> Wt[N][K] f16 (tiled transpose through LDS)
// ---------------------------------------------------------------------------
__global__ void k_transpose_to_f16(const float* __restrict__ W,
                                   _Float16* __restrict__ Wt, int K, int N) {
  __shared__ float tile[32][33];
  int k0 = blockIdx.y * 32, n0 = blockIdx.x * 32;
  int tx = threadIdx.x, ty = threadIdx.y;  // blockDim = (32, 8)
  for (int r = ty; r < 32; r += 8)
    tile[r][tx] = W[(size_t)(k0 + r) * N + n0 + tx];
  __syncthreads();
  for (int r = ty; r < 32; r += 8)
    Wt[(size_t)(n0 + r) * K + k0 + tx] = (_Float16)tile[tx][r];
}

// ---------------------------------------------------------------------------
// GEMM: C[M][N] f32 = A[M][K] f16 * Bt[N][K] f16 + bias.
// Block = 128 threads (4 waves). Block tile 64x64, each wave 16(M)x64(N).
// TDM path: wave 0 DMAs a 64-row x 64-half B tile (8 KB) into LDS per step,
// double buffered with s_wait_tensorcnt(1) pipelining; all 4 waves read B
// fragments from LDS (ds_load_b128), A fragments direct from global.
// ---------------------------------------------------------------------------
__global__ void __launch_bounds__(128)
k_gemm_f16_wmma(const _Float16* __restrict__ A, const _Float16* __restrict__ Bt,
                const float* __restrict__ bias, float* __restrict__ C,
                int M, int N, int K) {
  int wave  = threadIdx.x >> 5;
  int lane  = threadIdx.x & 31;
  int mbase = blockIdx.y * 64 + wave * 16;
  int nbase = blockIdx.x * 64;

  v8f acc[4] = {};

#if HAVE_TDM
  __shared__ __attribute__((aligned(64))) _Float16 Bs[2][64 * 64];  // 2x8KB
  unsigned bs_off[2];
  bs_off[0] = (unsigned)(uintptr_t)(void*)&Bs[0][0];  // flat[31:0] == LDS off
  bs_off[1] = (unsigned)(uintptr_t)(void*)&Bs[1][0];
  const bool issuer = (threadIdx.x < 32);  // wave 0 (uniform per wave)
  const int nsteps = K / 64;

  if (issuer)
    tdm_load_2d_f16(Bt + (size_t)nbase * K, bs_off[0], 64, 64,
                    (unsigned)K, (unsigned)N, (unsigned)K);

  for (int s = 0; s < nsteps; ++s) {
    int k0 = s * 64;
    if (issuer) {
      if (s + 1 < nsteps) {
        tdm_load_2d_f16(Bt + (size_t)nbase * K + k0 + 64, bs_off[(s + 1) & 1],
                        64, 64, (unsigned)K, (unsigned)N, (unsigned)K);
        __builtin_amdgcn_s_wait_tensorcnt(1);  // current buffer complete
      } else {
        __builtin_amdgcn_s_wait_tensorcnt(0);
      }
    }
    __syncthreads();  // publish LDS tile to all waves

    if (k0 + 64 < K)
      __builtin_prefetch(A + (size_t)mbase * K + k0 + 64, 0, 1);
    const _Float16* Bl = &Bs[s & 1][0];
#pragma unroll
    for (int ks = 0; ks < 2; ++ks) {
      v16h a = load_frag_A(A, K, mbase, k0 + ks * 32);
#pragma unroll
      for (int n = 0; n < 4; ++n) {
        v16h b = load_frag_B(Bl, 64, n * 16, ks * 32);
        acc[n] = __builtin_amdgcn_wmma_f32_16x16x32_f16(
            false, a, false, b, (short)0, acc[n], false, false);
      }
    }
    __syncthreads();  // all reads done before buffer reuse at step s+2
  }
#else
  for (int k0 = 0; k0 < K; k0 += 32) {
    if (k0 + 32 < K) {
      __builtin_prefetch(A + (size_t)mbase * K + k0 + 32, 0, 1);
      __builtin_prefetch(Bt + (size_t)nbase * K + k0 + 32, 0, 1);
    }
    v16h a = load_frag_A(A, K, mbase, k0);
#pragma unroll
    for (int n = 0; n < 4; ++n) {
      v16h b = load_frag_B(Bt, K, nbase + n * 16, k0);
      acc[n] = __builtin_amdgcn_wmma_f32_16x16x32_f16(
          false, a, false, b, (short)0, acc[n], false, false);
    }
  }
#endif

  int half = lane >> 4;
  int c0   = lane & 15;
#pragma unroll
  for (int n = 0; n < 4; ++n) {
    int col  = nbase + n * 16 + c0;
    float bv = bias ? bias[col] : 0.0f;
#pragma unroll
    for (int j = 0; j < 8; ++j) {
      int row = mbase + j + 8 * half;  // C layout: VGPR j -> row j / j+8
      C[(size_t)row * N + col] = acc[n][j] + bv;
    }
  }
}

// ---------------------------------------------------------------------------
// Per-(token, head) quant-dequant + layout shuffle.
// Block = 512 threads (16 waves), one block per token s, wave w = head h.
//   Qh[h][s][d]  = f16(Q * 0.125)                (1/sqrt(HD) folded in)
//   Kdq[h][s][d] = f16(quant-dequant 4-bit K)
//   Vt[h][d][s]  = f16(quant-dequant 8-bit V)    (transposed for PV B-frags)
// ---------------------------------------------------------------------------
__global__ void __launch_bounds__(512)
k_quant_pack(const float* __restrict__ Qf, const float* __restrict__ Kf,
             const float* __restrict__ Vf, _Float16* __restrict__ Qh,
             _Float16* __restrict__ Kdq, _Float16* __restrict__ Vt) {
  int s    = blockIdx.x;
  int h    = threadIdx.x >> 5;
  int lane = threadIdx.x & 31;
  int d0   = lane * 2;

  // Q: scale by 1/8, repack per-head
  const float* qrow = Qf + (size_t)s * D_DIM + h * HDIM;
  size_t qo = ((size_t)h * S_LEN + s) * HDIM + d0;
  Qh[qo]     = (_Float16)(qrow[d0] * 0.125f);
  Qh[qo + 1] = (_Float16)(qrow[d0 + 1] * 0.125f);

  // K: 4-bit asymmetric quant-dequant over head_dim
  {
    const float* krow = Kf + (size_t)s * D_DIM + h * HDIM;
    float a = krow[d0], b = krow[d0 + 1];
    float lo = fminf(a, b), hi = fmaxf(a, b);
#pragma unroll
    for (int m = 1; m < 32; m <<= 1) {
      lo = fminf(lo, __shfl_xor(lo, m, 32));
      hi = fmaxf(hi, __shfl_xor(hi, m, 32));
    }
    float scale = fmaxf((hi - lo) * (1.0f / 15.0f), 1e-8f);
    float inv   = 1.0f / scale;
    size_t ko = ((size_t)h * S_LEN + s) * HDIM + d0;
    Kdq[ko]     = (_Float16)(rintf((a - lo) * inv) * scale + lo);
    Kdq[ko + 1] = (_Float16)(rintf((b - lo) * inv) * scale + lo);
  }

  // V: 8-bit asymmetric quant-dequant, stored transposed [h][d][s]
  {
    const float* vrow = Vf + (size_t)s * D_DIM + h * HDIM;
    float a = vrow[d0], b = vrow[d0 + 1];
    float lo = fminf(a, b), hi = fmaxf(a, b);
#pragma unroll
    for (int m = 1; m < 32; m <<= 1) {
      lo = fminf(lo, __shfl_xor(lo, m, 32));
      hi = fmaxf(hi, __shfl_xor(hi, m, 32));
    }
    float scale = fmaxf((hi - lo) * (1.0f / 255.0f), 1e-8f);
    float inv   = 1.0f / scale;
    Vt[((size_t)h * HDIM + d0) * S_LEN + s] =
        (_Float16)(rintf((a - lo) * inv) * scale + lo);
    Vt[((size_t)h * HDIM + d0 + 1) * S_LEN + s] =
        (_Float16)(rintf((b - lo) * inv) * scale + lo);
  }
}

// ---------------------------------------------------------------------------
// Cycle embedding: emb[h][s] = tf[s]·Wc[:,h] + bc[h]; normalize over S;
// cq = cn * cycle_scale[h]; ck = cn.  One block (256 thr) per head.
// ---------------------------------------------------------------------------
__global__ void __launch_bounds__(256)
k_cycle(const float* __restrict__ tf, const float* __restrict__ Wc,
        const float* __restrict__ bc, const float* __restrict__ cscale,
        float* __restrict__ cq, float* __restrict__ ck) {
  int h = blockIdx.x;
  __shared__ float red[256];
  float wcol[TFDIM];
#pragma unroll
  for (int t = 0; t < TFDIM; ++t) wcol[t] = Wc[t * NHEAD + h];

  float sumsq = 0.0f;
  for (int s = threadIdx.x; s < S_LEN; s += 256) {
    float e = bc[h];
    const float* row = tf + (size_t)s * TFDIM;
#pragma unroll
    for (int t = 0; t < TFDIM; ++t) e += row[t] * wcol[t];
    ck[(size_t)h * S_LEN + s] = e;
    sumsq += e * e;
  }
  red[threadIdx.x] = sumsq;
  __syncthreads();
  for (int st = 128; st > 0; st >>= 1) {
    if (threadIdx.x < st) red[threadIdx.x] += red[threadIdx.x + st];
    __syncthreads();
  }
  float inv = 1.0f / fmaxf(sqrtf(red[0]), 1e-12f);
  float cs  = cscale[h];
  for (int s = threadIdx.x; s < S_LEN; s += 256) {
    float e = ck[(size_t)h * S_LEN + s] * inv;
    ck[(size_t)h * S_LEN + s] = e;
    cq[(size_t)h * S_LEN + s] = e * cs;
  }
}

// ---------------------------------------------------------------------------
// Flash attention per head. Grid = (S/64, H), block = 128 thr (4 waves).
// Each wave: 16 queries x full HD=64 output, streaming 64-key chunks.
// TDM path: wave 0 DMAs the shared K chunk (64 keys x 64 dims) and V chunk
// (64 dims x 64 keys from Vt) into LDS, double buffered with
// s_wait_tensorcnt(2); waves read fragments from LDS instead of 4x redundant
// global loads. scores = (Q*0.125) Kdq^T + cq[q]*ck[k]; online softmax;
// O += P Vdq. Output gathered as Og[s][h*64+d] f16 for the projection GEMM.
// ---------------------------------------------------------------------------
__global__ void __launch_bounds__(128)
k_flash_attn(const _Float16* __restrict__ Qh, const _Float16* __restrict__ Kdq,
             const _Float16* __restrict__ Vt, const float* __restrict__ cq,
             const float* __restrict__ ck, _Float16* __restrict__ Og) {
  int h     = blockIdx.y;
  int wave  = threadIdx.x >> 5;
  int lane  = threadIdx.x & 31;
  int half  = lane >> 4;
  int col   = lane & 15;
  int qbase = blockIdx.x * 64 + wave * 16;

  const _Float16* Qb = Qh + (size_t)h * S_LEN * HDIM;
  const _Float16* Kb = Kdq + (size_t)h * S_LEN * HDIM;
  const _Float16* Vb = Vt + (size_t)h * HDIM * S_LEN;

  v16h qa0 = load_frag_A(Qb, HDIM, qbase, 0);
  v16h qa1 = load_frag_A(Qb, HDIM, qbase, 32);

  float cqv[8];
#pragma unroll
  for (int j = 0; j < 8; ++j)
    cqv[j] = cq[(size_t)h * S_LEN + qbase + j + 8 * half];

  float m[8], l[8];
#pragma unroll
  for (int j = 0; j < 8; ++j) { m[j] = -INFINITY; l[j] = 0.0f; }
  v8f O[4] = {};

  __shared__ __attribute__((aligned(64))) _Float16 Plds[4][16 * 72];

#if HAVE_TDM
  __shared__ __attribute__((aligned(64))) _Float16 Ks[2][64 * 64];  // 2x8KB
  __shared__ __attribute__((aligned(64))) _Float16 Vs[2][64 * 64];  // 2x8KB
  unsigned ks_off[2], vs_off[2];
  ks_off[0] = (unsigned)(uintptr_t)(void*)&Ks[0][0];
  ks_off[1] = (unsigned)(uintptr_t)(void*)&Ks[1][0];
  vs_off[0] = (unsigned)(uintptr_t)(void*)&Vs[0][0];
  vs_off[1] = (unsigned)(uintptr_t)(void*)&Vs[1][0];
  const bool issuer = (threadIdx.x < 32);
  if (issuer) {
    // K chunk: 64 rows (keys) x 64 halves (dims), row stride HDIM
    tdm_load_2d_f16(Kb, ks_off[0], 64, 64, HDIM, S_LEN, HDIM);
    // V chunk: 64 rows (dims) x 64 halves (keys), row stride S_LEN
    tdm_load_2d_f16(Vb, vs_off[0], 64, 64, S_LEN, HDIM, S_LEN);
  }
#endif

  _Float16* myP = &Plds[wave][0];

  for (int kc = 0, step = 0; kc < S_LEN; kc += 64, ++step) {
#if HAVE_TDM
    if (issuer) {
      if (kc + 64 < S_LEN) {
        tdm_load_2d_f16(Kb + (size_t)(kc + 64) * HDIM, ks_off[(step + 1) & 1],
                        64, 64, HDIM, S_LEN, HDIM);
        tdm_load_2d_f16(Vb + (kc + 64), vs_off[(step + 1) & 1],
                        64, 64, S_LEN, HDIM, S_LEN);
        __builtin_amdgcn_s_wait_tensorcnt(2);  // current pair complete
      } else {
        __builtin_amdgcn_s_wait_tensorcnt(0);
      }
    }
    __syncthreads();  // publish K/V chunk to all waves
    const _Float16* Kl = &Ks[step & 1][0];
    const _Float16* Vl = &Vs[step & 1][0];
#endif

    // ---- scores: 4 tiles of 16q x 16k, K-dim = 64 head dims -------------
    v8f sc[4];
#pragma unroll
    for (int n = 0; n < 4; ++n) {
#if HAVE_TDM
      v16h b0 = load_frag_B(Kl, 64, n * 16, 0);
      v16h b1 = load_frag_B(Kl, 64, n * 16, 32);
#else
      v16h b0 = load_frag_B(Kb, HDIM, kc + n * 16, 0);
      v16h b1 = load_frag_B(Kb, HDIM, kc + n * 16, 32);
#endif
      v8f z = {};
      z = __builtin_amdgcn_wmma_f32_16x16x32_f16(false, qa0, false, b0,
                                                 (short)0, z, false, false);
      z = __builtin_amdgcn_wmma_f32_16x16x32_f16(false, qa1, false, b1,
                                                 (short)0, z, false, false);
      float ckv = ck[(size_t)h * S_LEN + kc + n * 16 + col];
#pragma unroll
      for (int j = 0; j < 8; ++j) z[j] += cqv[j] * ckv;  // rank-1 cycle bias
      sc[n] = z;
    }

    // ---- row max (rows live per-VGPR, cols per-lane within 16-lane half) -
    float mn[8], alpha[8];
#pragma unroll
    for (int j = 0; j < 8; ++j) {
      float v = fmaxf(fmaxf(sc[0][j], sc[1][j]), fmaxf(sc[2][j], sc[3][j]));
#pragma unroll
      for (int msk = 1; msk < 16; msk <<= 1)
        v = fmaxf(v, __shfl_xor(v, msk, 32));
      mn[j]    = fmaxf(m[j], v);
      alpha[j] = __expf(m[j] - mn[j]);
      m[j]     = mn[j];
    }

    // ---- exp, row sums, stage P to LDS (C-layout -> row-major) ----------
    float ls[8] = {0, 0, 0, 0, 0, 0, 0, 0};
#pragma unroll
    for (int n = 0; n < 4; ++n) {
#pragma unroll
      for (int j = 0; j < 8; ++j) {
        float p = __expf(sc[n][j] - mn[j]);
        ls[j] += p;
        myP[(j + 8 * half) * 72 + n * 16 + col] = (_Float16)p;
      }
    }
#pragma unroll
    for (int j = 0; j < 8; ++j) {
      float t = ls[j];
#pragma unroll
      for (int msk = 1; msk < 16; msk <<= 1) t += __shfl_xor(t, msk, 32);
      l[j] = l[j] * alpha[j] + t;
    }

    // ---- rescale O, then O += P * V -------------------------------------
#pragma unroll
    for (int t2 = 0; t2 < 4; ++t2)
#pragma unroll
      for (int j = 0; j < 8; ++j) O[t2][j] *= alpha[j];

    __syncthreads();  // P tile visible (also orders LDS w->r)
    v16h pa0 = load_frag_A(myP, 72, 0, 0);
    v16h pa1 = load_frag_A(myP, 72, 0, 32);
#pragma unroll
    for (int t2 = 0; t2 < 4; ++t2) {
#if HAVE_TDM
      v16h vb0 = load_frag_B(Vl, 64, t2 * 16, 0);
      v16h vb1 = load_frag_B(Vl, 64, t2 * 16, 32);
#else
      v16h vb0 = load_frag_B(Vb, S_LEN, t2 * 16, kc);
      v16h vb1 = load_frag_B(Vb, S_LEN, t2 * 16, kc + 32);
#endif
      O[t2] = __builtin_amdgcn_wmma_f32_16x16x32_f16(false, pa0, false, vb0,
                                                     (short)0, O[t2], false, false);
      O[t2] = __builtin_amdgcn_wmma_f32_16x16x32_f16(false, pa1, false, vb1,
                                                     (short)0, O[t2], false, false);
    }
    __syncthreads();  // reads done before P/K/V buffers are reused
  }

  // ---- finalize: O / l, gather into [s][h*64+d] f16 ----------------------
#pragma unroll
  for (int t2 = 0; t2 < 4; ++t2) {
#pragma unroll
    for (int j = 0; j < 8; ++j) {
      int q = qbase + j + 8 * half;
      Og[(size_t)q * D_DIM + h * HDIM + t2 * 16 + col] =
          (_Float16)(O[t2][j] / l[j]);
    }
  }
}

// ---------------------------------------------------------------------------
// Host-side orchestration
// ---------------------------------------------------------------------------
extern "C" void kernel_launch(void* const* d_in, const int* in_sizes, int n_in,
                              void* d_out, int out_size, void* d_ws,
                              size_t ws_size, hipStream_t stream) {
  const float* x  = (const float*)d_in[0];
  const float* tf = (const float*)d_in[1];
  // d_in[2] = mask (all true) -> unused
  const float* Wq = (const float*)d_in[3];
  const float* bq = (const float*)d_in[4];
  const float* Wk = (const float*)d_in[5];
  const float* bk = (const float*)d_in[6];
  const float* Wv = (const float*)d_in[7];
  const float* bv = (const float*)d_in[8];
  const float* Wo = (const float*)d_in[9];
  const float* bo = (const float*)d_in[10];
  const float* Wc = (const float*)d_in[11];
  const float* bc = (const float*)d_in[12];
  const float* cs = (const float*)d_in[13];

  char* wp = (char*)d_ws;
  auto carve = [&](size_t bytes) -> char* {
    char* p = wp;
    wp += (bytes + 255) & ~(size_t)255;
    return p;
  };
  const size_t SD = (size_t)S_LEN * D_DIM;
  _Float16* x16  = (_Float16*)carve(SD * 2);
  _Float16* WqT  = (_Float16*)carve((size_t)D_DIM * D_DIM * 2);
  _Float16* WkT  = (_Float16*)carve((size_t)D_DIM * D_DIM * 2);
  _Float16* WvT  = (_Float16*)carve((size_t)D_DIM * D_DIM * 2);
  _Float16* WoT  = (_Float16*)carve((size_t)D_DIM * D_DIM * 2);
  float*    Qf   = (float*)carve(SD * 4);
  float*    Kf   = (float*)carve(SD * 4);
  float*    Vf   = (float*)carve(SD * 4);
  _Float16* Qh   = (_Float16*)carve(SD * 2);
  _Float16* Kdq  = (_Float16*)carve(SD * 2);
  _Float16* Vt   = (_Float16*)carve(SD * 2);
  _Float16* Og   = (_Float16*)carve(SD * 2);
  float*    cqb  = (float*)carve((size_t)NHEAD * S_LEN * 4);
  float*    ckb  = (float*)carve((size_t)NHEAD * S_LEN * 4);
  (void)ws_size; (void)in_sizes; (void)n_in; (void)out_size;

  // 1) x -> f16
  k_f32_to_f16<<<2048, 256, 0, stream>>>(x, x16, (int)SD);

  // 2) weights -> transposed f16
  {
    dim3 g(D_DIM / 32, D_DIM / 32), b(32, 8);
    k_transpose_to_f16<<<g, b, 0, stream>>>(Wq, WqT, D_DIM, D_DIM);
    k_transpose_to_f16<<<g, b, 0, stream>>>(Wk, WkT, D_DIM, D_DIM);
    k_transpose_to_f16<<<g, b, 0, stream>>>(Wv, WvT, D_DIM, D_DIM);
    k_transpose_to_f16<<<g, b, 0, stream>>>(Wo, WoT, D_DIM, D_DIM);
  }

  // 3) Q/K/V projection GEMMs
  {
    dim3 g(D_DIM / 64, S_LEN / 64), b(128);
    k_gemm_f16_wmma<<<g, b, 0, stream>>>(x16, WqT, bq, Qf, S_LEN, D_DIM, D_DIM);
    k_gemm_f16_wmma<<<g, b, 0, stream>>>(x16, WkT, bk, Kf, S_LEN, D_DIM, D_DIM);
    k_gemm_f16_wmma<<<g, b, 0, stream>>>(x16, WvT, bv, Vf, S_LEN, D_DIM, D_DIM);
  }

  // 4) quant-dequant + per-head layouts
  k_quant_pack<<<S_LEN, 512, 0, stream>>>(Qf, Kf, Vf, Qh, Kdq, Vt);

  // 5) cycle embedding / normalization
  k_cycle<<<NHEAD, 256, 0, stream>>>(tf, Wc, bc, cs, cqb, ckb);

  // 6) flash attention
  {
    dim3 g(S_LEN / 64, NHEAD), b(128);
    k_flash_attn<<<g, b, 0, stream>>>(Qh, Kdq, Vt, cqb, ckb, Og);
  }

  // 7) output projection -> d_out (f32)
  {
    dim3 g(D_DIM / 64, S_LEN / 64), b(128);
    k_gemm_f16_wmma<<<g, b, 0, stream>>>(Og, WoT, bo, (float*)d_out, S_LEN,
                                         D_DIM, D_DIM);
  }
}
